// RGATBackbone_43387759624624
// MI455X (gfx1250) — matrix operand order repaired and checked by reading, not compiled, not run on hardware
//
#include <hip/hip_runtime.h>
#include <math.h>

typedef __attribute__((ext_vector_type(16))) __bf16 v16bf;
typedef __attribute__((ext_vector_type(8)))  __bf16 v8bf;
typedef __attribute__((ext_vector_type(8)))  float  v8f;
typedef __attribute__((ext_vector_type(4)))  float  v4f;

#define NN    50000
#define NE    600000
#define HID   128
#define HEADS 4
#define RELS  2
#define BSTR  136   // padded LDS row stride (bf16 elems): 272 B rotates banks by 4/row

// ---------------------------------------------------------------------------
// Weight prep: convert fp32 -> bf16; W is [r][k][o] row-major, we store
// Wt[l][r][o][k] (transposed) so B-fragments are contiguous per lane.
// pW is already [o][k] row-major (Linear weight), convert only.
// ---------------------------------------------------------------------------
__global__ void prep_weights(const float* __restrict__ W0, const float* __restrict__ W1,
                             const float* __restrict__ pW0, const float* __restrict__ pW1,
                             __bf16* __restrict__ Wt, __bf16* __restrict__ pWb)
{
    int i = blockIdx.x * blockDim.x + threadIdx.x;
    if (i < 2 * RELS * HID * HID) {
        int k = i % HID;
        int o = (i / HID) % HID;
        int r = (i / (HID * HID)) % RELS;
        int l = i / (RELS * HID * HID);
        const float* W = l ? W1 : W0;
        Wt[i] = (__bf16)W[((long)r * HID + k) * HID + o];
    }
    if (i < 2 * HID * HID) {
        int l = i / (HID * HID);
        const float* pW = l ? pW1 : pW0;
        pWb[i] = (__bf16)pW[i % (HID * HID)];
    }
}

// ---------------------------------------------------------------------------
// WMMA GEMM: C[rows x 128] = A[rows x 128] (fp32 -> bf16 on load) x B[128 x 128].
// B (32 KB bf16) is staged once per workgroup into LDS via the gfx1250 async
// global->LDS path (ASYNCcnt / s_wait_asynccnt), rows padded to kill bank
// conflicts. One wave per 16-row strip keeps 8 independent 16x16 accumulators
// (one per column tile) and loops k outermost, so the 8 WMMAs per k-step are
// independent chains the scheduler can pipeline against the ds_load_b128s.
// ---------------------------------------------------------------------------
__global__ void gemm_bf16_wmma(const float* __restrict__ A,
                               const __bf16* __restrict__ Bw,
                               float* __restrict__ C,
                               int rows, long Bstride, long Cstride,
                               const float* __restrict__ bias,
                               int apply_elu)
{
    __shared__ __bf16 Bs[HID * BSTR];

    const __bf16* Bp = Bw + (long)blockIdx.y * Bstride;
    float*        Cp = C  + (long)blockIdx.y * Cstride;

    // ---- async stage B into LDS (all threads; barrier before any exit) ----
    {
        // 128 rows x 256 B = 2048 b128 chunks; 8 chunks per thread.
#pragma unroll
        for (int j = 0; j < 8; ++j) {
            const int chunk = j * 256 + (int)threadIdx.x;
            const int row   = chunk >> 4;          // 16 chunks per row
            const int within = (chunk & 15) * 16;  // byte offset inside row
            const char* g = (const char*)Bp + row * (HID * 2) + within;
            unsigned lds_addr =
                (unsigned)(unsigned long long)((char*)Bs + row * (BSTR * 2) + within);
            asm volatile("global_load_async_to_lds_b128 %0, %1, off"
                         :: "v"(lds_addr), "v"(g) : "memory");
        }
        asm volatile("s_wait_asynccnt 0" ::: "memory");
    }
    __syncthreads();

    const int lane = threadIdx.x & 31;
    const int wave = threadIdx.x >> 5;
    const int mtile  = blockIdx.x * 8 + wave;
    const int mtiles = (rows + 15) >> 4;
    if (mtile >= mtiles) return;            // wave-uniform -> EXEC all-1s for WMMA

    const int half = lane >> 4;             // 0: lanes 0-15, 1: lanes 16-31
    const int l16  = lane & 15;

    int rowA = mtile * 16 + l16;
    if (rowA >= rows) rowA = rows - 1;      // clamp (rows%16==0 here anyway)
    const float* arow = A + (long)rowA * HID;

    // A fragments: ISA 16-bit A 16x32 layout => lane holds K in
    // [kb, kb+8) and [kb+16, kb+24) with kb = ks*32 + half*8.
    v16bf afrag[4];
#pragma unroll
    for (int ks = 0; ks < 4; ++ks) {
        const int kb = ks * 32 + half * 8;
        v4f f0 = *(const v4f*)(arow + kb);
        v4f f1 = *(const v4f*)(arow + kb + 4);
        v4f f2 = *(const v4f*)(arow + kb + 16);
        v4f f3 = *(const v4f*)(arow + kb + 20);
#pragma unroll
        for (int i = 0; i < 4; ++i) {
            afrag[ks][i]      = (__bf16)f0[i];
            afrag[ks][4 + i]  = (__bf16)f1[i];
            afrag[ks][8 + i]  = (__bf16)f2[i];
            afrag[ks][12 + i] = (__bf16)f3[i];
        }
    }

    // 8 independent accumulator chains (one per 16-wide column tile).
    v8f acc[8];
#pragma unroll
    for (int ct = 0; ct < 8; ++ct) acc[ct] = (v8f){};

    // Per-lane LDS base for this lane's B column within tile ct:
    // B 32x16 layout: lane = column N, lane holds K in [kb, kb+8) and
    // [kb+16, kb+24) with kb = half*8 (matches A, [o][k] LDS image).
#pragma unroll
    for (int ks = 0; ks < 4; ++ks) {
#pragma unroll
        for (int ct = 0; ct < 8; ++ct) {
            const __bf16* bp = &Bs[(ct * 16 + l16) * BSTR + half * 8 + ks * 32];
            v8bf b0 = *(const v8bf*)(bp);
            v8bf b1 = *(const v8bf*)(bp + 16);
            v16bf bfrag;
#pragma unroll
            for (int i = 0; i < 8; ++i) { bfrag[i] = b0[i]; bfrag[8 + i] = b1[i]; }
            acc[ct] = __builtin_amdgcn_wmma_f32_16x16x32_bf16(
                false, afrag[ks], false, bfrag, (short)0, acc[ct], false, false);
        }
    }

    // C/D layout: VGPR v holds row (v + 8*half), column = l16 within tile.
#pragma unroll
    for (int ct = 0; ct < 8; ++ct) {
        const int col = ct * 16 + l16;
        const float b = bias ? bias[col] : 0.0f;
#pragma unroll
        for (int v = 0; v < 8; ++v) {
            const int row = mtile * 16 + half * 8 + v;
            if (row < rows) {
                float val = acc[ct][v] + b;
                if (apply_elu) val = (val > 0.0f) ? val : (expf(val) - 1.0f);
                Cp[(long)row * HID + col] = val;
            }
        }
    }
}

// ---------------------------------------------------------------------------
// Per-node attention projections: qs[r,n,h] = xw[r,n,:] . Q[:,h], same for K.
// ---------------------------------------------------------------------------
__global__ void node_qk(const float* __restrict__ xw,
                        const float* __restrict__ Q, const float* __restrict__ K,
                        float* __restrict__ qs, float* __restrict__ ks)
{
    int i = blockIdx.x * blockDim.x + threadIdx.x;   // over R*N*HEADS
    if (i >= RELS * NN * HEADS) return;
    const int h  = i & (HEADS - 1);
    const long rn = i >> 2;
    const float* row = xw + rn * HID;
    float aq = 0.0f, ak = 0.0f;
#pragma unroll 8
    for (int o = 0; o < HID; ++o) {
        const float v = row[o];
        aq += v * Q[o * HEADS + h];
        ak += v * K[o * HEADS + h];
    }
    qs[i] = aq;
    ks[i] = ak;
}

__global__ void init_buffers(float* __restrict__ m, float* __restrict__ s,
                             float* __restrict__ agg)
{
    int i = blockIdx.x * blockDim.x + threadIdx.x;
    if (i < NN * HID) agg[i] = 0.0f;
    if (i < NN * HEADS) { m[i] = -INFINITY; s[i] = 0.0f; }
}

__device__ __forceinline__ float atomicMaxFloat(float* addr, float val)
{
    if (val >= 0.0f)
        return __int_as_float(atomicMax((int*)addr, __float_as_int(val)));
    return __uint_as_float(atomicMin((unsigned int*)addr, __float_as_uint(val)));
}

// Pass 1: alpha = leaky_relu(q[dst] + k[src]); segment max over dst.
__global__ void edge_alpha_max(const int* __restrict__ eidx, const int* __restrict__ etype,
                               const float* __restrict__ qs, const float* __restrict__ ks,
                               float* __restrict__ alpha, float* __restrict__ m)
{
    int i = blockIdx.x * blockDim.x + threadIdx.x;   // over E*HEADS
    if (i >= NE * HEADS) return;
    const int h = i & (HEADS - 1);
    const int e = i >> 2;
    const int s = eidx[e];
    const int d = eidx[NE + e];
    const int r = etype[e];
    float a = qs[((long)r * NN + d) * HEADS + h] + ks[((long)r * NN + s) * HEADS + h];
    a = (a > 0.0f) ? a : 0.2f * a;
    alpha[i] = a;
    atomicMaxFloat(&m[(long)d * HEADS + h], a);
}

// Pass 2: e = exp(alpha - m[dst]); segment sum over dst.
__global__ void edge_exp_sum(const int* __restrict__ eidx,
                             float* __restrict__ alpha,
                             const float* __restrict__ m, float* __restrict__ s)
{
    int i = blockIdx.x * blockDim.x + threadIdx.x;
    if (i >= NE * HEADS) return;
    const int h = i & (HEADS - 1);
    const int e = i >> 2;
    const int d = eidx[NE + e];
    float mm = m[(long)d * HEADS + h];
    if (mm == -INFINITY) mm = 0.0f;
    const float ev = expf(alpha[i] - mm);
    alpha[i] = ev;
    atomicAdd(&s[(long)d * HEADS + h], ev);
}

// Pass 3: one wave per edge; lane handles 4 consecutive channels (one head),
// float4 gather from L2-resident xw, 4 atomic adds into agg[dst].
__global__ void edge_scatter(const int* __restrict__ eidx, const int* __restrict__ etype,
                             const float* __restrict__ alpha, const float* __restrict__ ssum,
                             const float* __restrict__ xw, float* __restrict__ agg)
{
    const int lane = threadIdx.x & 31;
    const int e = (blockIdx.x * blockDim.x + threadIdx.x) >> 5;
    if (e >= NE) return;
    const int s = eidx[e];
    const int d = eidx[NE + e];
    const int r = etype[e];
    const int head = lane >> 3;                       // 4*lane .. 4*lane+3 share a head
    const float a = alpha[(long)e * HEADS + head] /
                    (ssum[(long)d * HEADS + head] + 1e-16f);
    const float* srcrow = xw + ((long)r * NN + s) * HID + lane * 4;
    const v4f v = *(const v4f*)srcrow;
    float* dstrow = agg + (long)d * HID + lane * 4;
#pragma unroll
    for (int j = 0; j < 4; ++j) atomicAdd(dstrow + j, v[j] * a);
}

__global__ void add_bias(float* __restrict__ agg, const float* __restrict__ cb)
{
    int i = blockIdx.x * blockDim.x + threadIdx.x;
    if (i < NN * HID) agg[i] += cb[i & (HID - 1)];
}

// ---------------------------------------------------------------------------
extern "C" void kernel_launch(void* const* d_in, const int* in_sizes, int n_in,
                              void* d_out, int out_size, void* d_ws, size_t ws_size,
                              hipStream_t stream)
{
    (void)in_sizes; (void)n_in; (void)out_size; (void)ws_size;

    const float* x     = (const float*)d_in[0];
    const int*   eidx  = (const int*)d_in[1];
    const int*   etype = (const int*)d_in[2];
    const float* Wl[2]  = { (const float*)d_in[3],  (const float*)d_in[9]  };
    const float* Ql[2]  = { (const float*)d_in[4],  (const float*)d_in[10] };
    const float* Kl[2]  = { (const float*)d_in[5],  (const float*)d_in[11] };
    const float* cbl[2] = { (const float*)d_in[6],  (const float*)d_in[12] };
    const float* pWl[2] = { (const float*)d_in[7],  (const float*)d_in[13] };
    const float* pbl[2] = { (const float*)d_in[8],  (const float*)d_in[14] };

    size_t off = 0;
    auto wsAlloc = [&](size_t bytes) -> void* {
        void* p = (void*)((char*)d_ws + off);
        off += (bytes + 255) & ~(size_t)255;
        return p;
    };
    float*  xw    = (float*)wsAlloc((size_t)RELS * NN * HID * 4);   // 51.2 MB
    float*  agg   = (float*)wsAlloc((size_t)NN * HID * 4);          // 25.6 MB
    float*  hbuf  = (float*)wsAlloc((size_t)NN * HID * 4);          // 25.6 MB
    float*  alpha = (float*)wsAlloc((size_t)NE * HEADS * 4);        //  9.6 MB
    float*  qs    = (float*)wsAlloc((size_t)RELS * NN * HEADS * 4); //  1.6 MB
    float*  ks    = (float*)wsAlloc((size_t)RELS * NN * HEADS * 4); //  1.6 MB
    float*  m     = (float*)wsAlloc((size_t)NN * HEADS * 4);        //  0.8 MB
    float*  ssum  = (float*)wsAlloc((size_t)NN * HEADS * 4);        //  0.8 MB
    __bf16* Wt    = (__bf16*)wsAlloc((size_t)2 * RELS * HID * HID * 2);
    __bf16* pWb   = (__bf16*)wsAlloc((size_t)2 * HID * HID * 2);

    const int mtiles   = (NN + 15) / 16;              // 3125
    const int gemmBlkX = (mtiles + 7) / 8;            // 8 waves (16-row strips)/block

    prep_weights<<<(2 * RELS * HID * HID + 255) / 256, 256, 0, stream>>>(
        Wl[0], Wl[1], pWl[0], pWl[1], Wt, pWb);

    const float* hin = x;
    float* hout[2] = { hbuf, (float*)d_out };

    for (int l = 0; l < 2; ++l) {
        // xw[r] = h @ W[r]   (bf16 WMMA, fp32 accumulate), one grid.y per relation
        gemm_bf16_wmma<<<dim3(gemmBlkX, RELS), 256, 0, stream>>>(
            hin, Wt + (size_t)l * RELS * HID * HID, xw,
            NN, (long)HID * HID, (long)NN * HID, nullptr, 0);

        node_qk<<<(RELS * NN * HEADS + 255) / 256, 256, 0, stream>>>(
            xw, Ql[l], Kl[l], qs, ks);

        init_buffers<<<(NN * HID + 255) / 256, 256, 0, stream>>>(m, ssum, agg);

        edge_alpha_max<<<(NE * HEADS + 255) / 256, 256, 0, stream>>>(
            eidx, etype, qs, ks, alpha, m);

        edge_exp_sum<<<(NE * HEADS + 255) / 256, 256, 0, stream>>>(
            eidx, alpha, m, ssum);

        edge_scatter<<<(NE * 32 + 255) / 256, 256, 0, stream>>>(
            eidx, etype, alpha, ssum, xw, agg);

        add_bias<<<(NN * HID + 255) / 256, 256, 0, stream>>>(agg, cbl[l]);

        // h = elu(hconv @ pW.T + pb)  (fused bias + ELU epilogue)
        gemm_bf16_wmma<<<dim3(gemmBlkX, 1), 256, 0, stream>>>(
            agg, pWb + (size_t)l * HID * HID, hout[l],
            NN, 0, 0, pbl[l], 1);

        hin = hbuf;
    }
}
